// SetAbstraction_63471026700451
// MI455X (gfx1250) — matrix operand order repaired and checked by reading, not compiled, not run on hardware
//
#include <hip/hip_runtime.h>
#include <hip/hip_bf16.h>

typedef _Float16 v16h __attribute__((ext_vector_type(16)));
typedef _Float16 h8   __attribute__((ext_vector_type(8)));
typedef _Float16 h2   __attribute__((ext_vector_type(2)));
typedef float    v8f  __attribute__((ext_vector_type(8)));

#define N_PTS   8192
#define KNBR    32
#define C_IN    64
#define C_MID   64
#define C_OUT   128
#define XS      104   // x column stride (halves), >=96, mult of 8
#define HS      72    // h1 column stride (halves), >=64, mult of 8
#define OS      132   // out staging row stride (floats)

__device__ __forceinline__ v16h ld2(const _Float16* p0, const _Float16* p1) {
    union { v16h v; h8 h[2]; } u;
    u.h[0] = *(const h8*)p0;
    u.h[1] = *(const h8*)p1;
    return u.v;
}

__global__ __launch_bounds__(256)
void sa_fused_kernel(const float* __restrict__ dp, const float* __restrict__ fj,
                     const float* __restrict__ W1, const float* __restrict__ b1,
                     const float* __restrict__ W2, const float* __restrict__ b2,
                     float* __restrict__ out)
{
    __shared__ _Float16 sW1[64 * 96];      // W1 padded 67->96, f16, row-major
    __shared__ _Float16 sW2[128 * 64];     // W2 f16, row-major
    __shared__ float    sB1[64];
    __shared__ float    sB2[128];
    __shared__ _Float16 sX [8][32 * XS];   // per-wave x tile, [k][c] f16
    __shared__ _Float16 sH1[8][32 * HS];   // per-wave h1 tile, [k][och] f16
    __shared__ float    sOut[32 * OS];     // [n_local][och] f32

    const int tid  = threadIdx.x;
    const int lane = tid & 31;
    const int w    = tid >> 5;
    const int l16  = lane & 15;
    const int hl   = lane >> 4;

    const int wg = blockIdx.x;
    const int b  = wg >> 8;            // 4 batches * 256 wgs each
    const int n0 = (wg & 255) * 32;    // 32 points per workgroup

    // ---- stage weights/biases into LDS (f16) ----
    for (int i = tid; i < 64 * 96; i += 256) {
        int r = i / 96, c = i - r * 96;
        sW1[i] = (c < 67) ? (_Float16)W1[r * 67 + c] : (_Float16)0.f;
    }
    for (int i = tid; i < 128 * 64; i += 256) sW2[i] = (_Float16)W2[i];
    if (tid < 64)  sB1[tid] = b1[tid];
    if (tid < 128) sB2[tid] = b2[tid];
    __syncthreads();

    // zero-fill padded channel region of this wave's x tile (c = 68..95), once
    {
        unsigned long long* zp = (unsigned long long*)&sX[w][lane * XS + 68];
        #pragma unroll
        for (int j = 0; j < 7; ++j) zp[j] = 0ull;
    }

    const size_t CS = (size_t)N_PTS * KNBR; // channel stride in dp/fj

    _Float16* xp  = sX[w];
    _Float16* h1p = sH1[w];

    #pragma unroll 1
    for (int it = 0; it < 4; ++it) {
        const int n  = n0 + w * 4 + it;
        const int pt = w * 4 + it;

        // ---- stage x = concat(dp, fj) column for k = lane, f32 -> f16 ----
        {
            const size_t baseD = ((size_t)b * 3  * N_PTS + n) * KNBR + lane;
            const size_t baseF = ((size_t)b * 64 * N_PTS + n) * KNBR + lane;
            _Float16* xr = xp + lane * XS;
            float d0 = dp[baseD], d1 = dp[baseD + CS], d2 = dp[baseD + 2 * CS];
            float f0 = fj[baseF];
            h2 t0; t0[0] = (_Float16)d0; t0[1] = (_Float16)d1;
            h2 t1; t1[0] = (_Float16)d2; t1[1] = (_Float16)f0;
            *(h2*)(xr + 0) = t0;
            *(h2*)(xr + 2) = t1;
            #pragma unroll
            for (int j = 1; j < 63; j += 2) {
                float a = fj[baseF + (size_t)j * CS];
                float c = fj[baseF + (size_t)(j + 1) * CS];
                h2 t; t[0] = (_Float16)a; t[1] = (_Float16)c;
                *(h2*)(xr + 3 + j) = t;                 // x-channel 3+j (even)
            }
            float fl = fj[baseF + 63 * CS];
            h2 t; t[0] = (_Float16)fl; t[1] = (_Float16)0.f;
            *(h2*)(xr + 66) = t;                        // covers c=66,67
        }

        // ---- layer 1: D1[och][k] = relu(W1 * x + b1), f16 WMMA, f32 acc ----
        #pragma unroll
        for (int t = 0; t < 2; ++t) {                   // column tiles (k)
            const int kcol = t * 16 + l16;
            v16h Bf[3];
            #pragma unroll
            for (int q = 0; q < 3; ++q) {
                const _Float16* p = xp + kcol * XS + q * 32 + hl * 8;
                Bf[q] = ld2(p, p + 16);
            }
            #pragma unroll
            for (int m = 0; m < 4; ++m) {               // och tiles
                const int ob = m * 16 + hl * 8;
                const float4* bp = (const float4*)&sB1[ob];
                float4 x0 = bp[0], x1 = bp[1];
                v8f acc;
                acc[0] = x0.x; acc[1] = x0.y; acc[2] = x0.z; acc[3] = x0.w;
                acc[4] = x1.x; acc[5] = x1.y; acc[6] = x1.z; acc[7] = x1.w;
                #pragma unroll
                for (int q = 0; q < 3; ++q) {
                    const _Float16* ap = sW1 + (m * 16 + l16) * 96 + q * 32 + hl * 8;
                    v16h A = ld2(ap, ap + 16);
                    acc = __builtin_amdgcn_wmma_f32_16x16x32_f16(
                              false, A, false, Bf[q], (short)0, acc, false, false);
                }
                h8 hv;
                #pragma unroll
                for (int r = 0; r < 8; ++r) hv[r] = (_Float16)fmaxf(acc[r], 0.f);
                *(h8*)(h1p + kcol * HS + ob) = hv;
            }
        }

        // ---- layer 2 (transposed): D2[k][och] = h1^T * W2^T + b2; max over k ----
        float vmax[8];
        #pragma unroll
        for (int i = 0; i < 8; ++i) vmax[i] = 0.f;      // relu folded into max

        #pragma unroll
        for (int kt = 0; kt < 2; ++kt) {
            const int krow = kt * 16 + l16;
            const _Float16* a0 = h1p + krow * HS + hl * 8;
            const _Float16* a1 = h1p + krow * HS + 32 + hl * 8;
            v16h A0 = ld2(a0, a0 + 16);
            v16h A1 = ld2(a1, a1 + 16);
            #pragma unroll
            for (int nt = 0; nt < 8; ++nt) {            // och tiles
                const float bb = sB2[nt * 16 + l16];
                v8f acc;
                #pragma unroll
                for (int r = 0; r < 8; ++r) acc[r] = bb;
                const _Float16* bq0 = sW2 + (nt * 16 + l16) * 64 + hl * 8;
                const _Float16* bq1 = bq0 + 32;
                v16h B0 = ld2(bq0, bq0 + 16);
                acc = __builtin_amdgcn_wmma_f32_16x16x32_f16(
                          false, A0, false, B0, (short)0, acc, false, false);
                v16h B1 = ld2(bq1, bq1 + 16);
                acc = __builtin_amdgcn_wmma_f32_16x16x32_f16(
                          false, A1, false, B1, (short)0, acc, false, false);
                float m = acc[0];
                #pragma unroll
                for (int r = 1; r < 8; ++r) m = fmaxf(m, acc[r]);
                m = fmaxf(m, __shfl_xor(m, 16, 32));    // join row halves
                vmax[nt] = fmaxf(vmax[nt], m);
            }
        }

        if (hl == 0) {
            #pragma unroll
            for (int nt = 0; nt < 8; ++nt)
                sOut[pt * OS + nt * 16 + l16] = vmax[nt];
        }
    }

    __syncthreads();

    // ---- coalesced write-out: out[b][och][n0..n0+31] ----
    {
        const int och = tid >> 1;          // 0..127
        const int nh  = tid & 1;           // n-half: 0..15 / 16..31
        float* op = out + ((size_t)(b * 128 + och)) * N_PTS + n0 + nh * 16;
        #pragma unroll
        for (int j = 0; j < 4; ++j) {
            float4 o;
            o.x = sOut[(nh * 16 + j * 4 + 0) * OS + och];
            o.y = sOut[(nh * 16 + j * 4 + 1) * OS + och];
            o.z = sOut[(nh * 16 + j * 4 + 2) * OS + och];
            o.w = sOut[(nh * 16 + j * 4 + 3) * OS + och];
            *(float4*)(op + j * 4) = o;
        }
    }
}

extern "C" void kernel_launch(void* const* d_in, const int* in_sizes, int n_in,
                              void* d_out, int out_size, void* d_ws, size_t ws_size,
                              hipStream_t stream) {
    // setup_inputs order: p, f, dp, fj, W1, b1, W2, b2  (p and f are unused)
    const float* dp = (const float*)d_in[2];
    const float* fj = (const float*)d_in[3];
    const float* W1 = (const float*)d_in[4];
    const float* b1 = (const float*)d_in[5];
    const float* W2 = (const float*)d_in[6];
    const float* b2 = (const float*)d_in[7];
    float* out = (float*)d_out;

    dim3 grid(1024);   // (B*N)/32 workgroups, 32 points each
    dim3 block(256);   // 8 waves (wave32)
    sa_fused_kernel<<<grid, block, 0, stream>>>(dp, fj, W1, b1, W2, b2, out);
    (void)in_sizes; (void)n_in; (void)out_size; (void)d_ws; (void)ws_size;
}
